// Causal_Transformer_11613591568642
// MI455X (gfx1250) — compile-verified, matrix operand order checked by reading, not run on hardware
//
#include <hip/hip_runtime.h>
#include <hip/hip_bf16.h>
#include <cstdint>
#include <cstddef>

// Problem constants (match reference)
#define B_    4
#define S_    1024
#define H_    1024
#define NH_   16
#define DK_   64
#define L_    2
#define MLP_  4096
#define HQKV_ 3072
#define MTOT_ (B_ * S_)   // 4096 rows

typedef __attribute__((ext_vector_type(16))) __bf16 v16bf;
typedef __attribute__((ext_vector_type(8)))  float  v8f;

union FragAB { v16bf v; uint4 q[2]; };

__device__ __forceinline__ __bf16 f32_to_bf16(float f) {
  uint32_t u = __float_as_uint(f);
  u += 0x7FFFu + ((u >> 16) & 1u);           // round-to-nearest-even
  uint16_t h = (uint16_t)(u >> 16);
  return __builtin_bit_cast(__bf16, h);
}
__device__ __forceinline__ float bf16_to_f32(__bf16 x) {
  uint16_t h = __builtin_bit_cast(uint16_t, x);
  return __uint_as_float(((uint32_t)h) << 16);
}

// CDNA5 async global->LDS copy: 64B per lane (4 x b128), contiguous in both
// spaces (ISA 10.7: INST_OFFSET is added to both the global and LDS address).
// Tracked by ASYNCcnt; must s_wait_asynccnt 0 before the consuming barrier.
__device__ __forceinline__ void async_ldx4_b128(const __bf16* g, const __bf16* l) {
  uint32_t la = (uint32_t)(uintptr_t)l;            // low 32 bits = LDS address
  uint64_t ga = (uint64_t)(uintptr_t)g;
  asm volatile(
      "global_load_async_to_lds_b128 %0, %1, off\n\t"
      "global_load_async_to_lds_b128 %0, %1, off offset:16\n\t"
      "global_load_async_to_lds_b128 %0, %1, off offset:32\n\t"
      "global_load_async_to_lds_b128 %0, %1, off offset:48"
      :: "v"(la), "v"(ga) : "memory");
}
__device__ __forceinline__ void wait_async0() {
  asm volatile("s_wait_asynccnt 0" ::: "memory");
}

// ---------------------------------------------------------------- utilities
__global__ __launch_bounds__(256)
void k_f32_to_bf16(const float* __restrict__ s, __bf16* __restrict__ d) {
  int i = blockIdx.x * 256 + threadIdx.x;
  float4 v = ((const float4*)s)[i];
  union { __bf16 b[4]; uint2 u; } o;
  o.b[0] = f32_to_bf16(v.x); o.b[1] = f32_to_bf16(v.y);
  o.b[2] = f32_to_bf16(v.z); o.b[3] = f32_to_bf16(v.w);
  ((uint2*)d)[i] = o.u;
}

__global__ __launch_bounds__(256)
void k_copy_f32(const float* __restrict__ s, float* __restrict__ d) {
  int i = blockIdx.x * 256 + threadIdx.x;
  ((float4*)d)[i] = ((const float4*)s)[i];
}

// ---------------------------------------------------------------- LayerNorm
__global__ __launch_bounds__(256)
void ln_kernel(const float* __restrict__ x, const float* __restrict__ g,
               const float* __restrict__ bta, __bf16* __restrict__ y)
{
  const int row = blockIdx.x;
  const int tid = threadIdx.x;
  const float4 xv = ((const float4*)(x + (size_t)row * H_))[tid];
  float s  = xv.x + xv.y + xv.z + xv.w;
  float s2 = xv.x*xv.x + xv.y*xv.y + xv.z*xv.z + xv.w*xv.w;
  #pragma unroll
  for (int d = 1; d < 32; d <<= 1) {
    s  += __shfl_xor(s,  d, 32);
    s2 += __shfl_xor(s2, d, 32);
  }
  __shared__ float rs[8], rq[8];
  __shared__ float mv[2];
  if ((tid & 31) == 0) { rs[tid >> 5] = s; rq[tid >> 5] = s2; }
  __syncthreads();
  if (tid == 0) {
    float a = 0.f, a2 = 0.f;
    #pragma unroll
    for (int i = 0; i < 8; ++i) { a += rs[i]; a2 += rq[i]; }
    float mean = a * (1.0f / (float)H_);
    float var  = a2 * (1.0f / (float)H_) - mean * mean;
    mv[0] = mean; mv[1] = rsqrtf(var + 1e-5f);
  }
  __syncthreads();
  const float mean = mv[0], rstd = mv[1];
  const float4 gv = ((const float4*)g)[tid];
  const float4 bv = ((const float4*)bta)[tid];
  union { __bf16 b[4]; uint2 u; } o;
  o.b[0] = f32_to_bf16((xv.x - mean) * rstd * gv.x + bv.x);
  o.b[1] = f32_to_bf16((xv.y - mean) * rstd * gv.y + bv.y);
  o.b[2] = f32_to_bf16((xv.z - mean) * rstd * gv.z + bv.z);
  o.b[3] = f32_to_bf16((xv.w - mean) * rstd * gv.w + bv.w);
  ((uint2*)(y + (size_t)row * H_))[tid] = o.u;
}

// ---------------------------------------------------------------- RoPE (in place on bf16 qkv)
__global__ __launch_bounds__(256)
void rope_kernel(__bf16* __restrict__ qkv, const int* __restrict__ pos)
{
  int idx = blockIdx.x * 256 + threadIdx.x;   // B*S*2*NH*32 threads
  int d  = idx & 31; int t = idx >> 5;
  int hh = t & 15;   t >>= 4;
  int qk = t & 1;    t >>= 1;
  int s  = t & 1023; int b = t >> 10;
  float invf  = __expf(-0.28782313662425572f * (float)d);  // ln(10000)/32
  float theta = (float)pos[s] * invf;
  float sn, cs; __sincosf(theta, &sn, &cs);
  size_t base = ((size_t)(b * S_ + s)) * HQKV_ + qk * H_ + hh * DK_ + d;
  float lo = bf16_to_f32(qkv[base]);
  float hi = bf16_to_f32(qkv[base + 32]);
  qkv[base]      = f32_to_bf16(lo * cs - hi * sn);
  qkv[base + 32] = f32_to_bf16(hi * cs + lo * sn);
}

// ---------------------------------------------------------------- WMMA GEMM
// C[M,N] = epi(A[M,K](bf16) @ W[K,N](bf16) + bias[N])
// EPI 0: store bf16 | EPI 1: gelu(tanh) -> bf16 | EPI 2: fp32 += (residual)
// 128x64 block tile, K-step 64, double-buffered LDS:
//   A tiles streamed with GLOBAL_LOAD_ASYNC_TO_LDS_B128 (ASYNCcnt engine),
//   B tiles transposed through VGPRs (async path cannot transpose).
#define GBK 64
#define GLD 72    // padded LDS stride: 144B = 16B-aligned, bank-rotating

template<int EPI>
__global__ __launch_bounds__(256)
void gemm_bf16(const __bf16* __restrict__ A, const __bf16* __restrict__ Bw,
               const float* __restrict__ bias, void* __restrict__ Cout,
               int N, int K)
{
  __shared__ __align__(16) __bf16 sA[2][128 * GLD];  // 2 x 18432 B
  __shared__ __align__(16) __bf16 sB[2][ 64 * GLD];  // 2 x  9216 B (transposed [n][k])

  const int tid  = threadIdx.x;
  const int lane = tid & 31;
  const int wave = tid >> 5;
  const int wm   = wave >> 1;           // 0..3 : wave M offset (x32)
  const int wn   = wave & 1;            // 0..1 : wave N offset (x32)
  const int half = lane >> 4;
  const int l16  = lane & 15;
  const int bm   = blockIdx.y * 128;
  const int bn   = blockIdx.x * 64;

  v8f acc[2][2] = {};

  const int ar   = tid >> 1;            // A tile row 0..127
  const int ac   = (tid & 1) * 32;      // A tile col {0,32}
  const int bk   = tid >> 2;            // B tile k 0..63
  const int bn16 = (tid & 3) * 16;      // B tile n {0,16,32,48}

  const __bf16* aptr = A  + (size_t)(bm + ar) * K + ac;
  const __bf16* bptr = Bw + (size_t)bk * N + bn + bn16;

  // ---- prologue: stage 0
  int cur = 0;
  async_ldx4_b128(aptr, &sA[0][ar * GLD + ac]);
  {
    uint4 b0 = *(const uint4*)(bptr);
    uint4 b1 = *(const uint4*)(bptr + 8);
    const __bf16* e0 = (const __bf16*)&b0;
    const __bf16* e1 = (const __bf16*)&b1;
    #pragma unroll
    for (int j = 0; j < 8; ++j) {
      sB[0][(bn16 + j)     * GLD + bk] = e0[j];
      sB[0][(bn16 + 8 + j) * GLD + bk] = e1[j];
    }
  }
  wait_async0();
  __syncthreads();

  for (int k0 = 0; k0 < K; k0 += GBK) {
    const int  nxt      = cur ^ 1;
    const bool has_next = (k0 + GBK) < K;

    uint4 nb0 = {0,0,0,0}, nb1 = {0,0,0,0};
    if (has_next) {
      async_ldx4_b128(aptr + k0 + GBK, &sA[nxt][ar * GLD + ac]);
      nb0 = *(const uint4*)(bptr + (size_t)(k0 + GBK) * N);
      nb1 = *(const uint4*)(bptr + (size_t)(k0 + GBK) * N + 8);
    }

    // fragments from current stage
    FragAB af[2][2], bfr[2][2];
    #pragma unroll
    for (int mt = 0; mt < 2; ++mt)
      #pragma unroll
      for (int ks = 0; ks < 2; ++ks) {
        const __bf16* p = &sA[cur][(wm*32 + mt*16 + l16) * GLD + ks*32 + half*8];
        af[mt][ks].q[0] = *(const uint4*)p;
        af[mt][ks].q[1] = *(const uint4*)(p + 16);
      }
    #pragma unroll
    for (int nt = 0; nt < 2; ++nt)
      #pragma unroll
      for (int ks = 0; ks < 2; ++ks) {
        const __bf16* p = &sB[cur][(wn*32 + nt*16 + l16) * GLD + ks*32 + half*8];
        bfr[nt][ks].q[0] = *(const uint4*)p;
        bfr[nt][ks].q[1] = *(const uint4*)(p + 16);
      }
    #pragma unroll
    for (int ks = 0; ks < 2; ++ks)
      #pragma unroll
      for (int mt = 0; mt < 2; ++mt)
        #pragma unroll
        for (int nt = 0; nt < 2; ++nt)
          acc[mt][nt] = __builtin_amdgcn_wmma_f32_16x16x32_bf16(
              false, af[mt][ks].v, false, bfr[nt][ks].v, (short)0,
              acc[mt][nt], false, false);

    if (has_next) {
      const __bf16* e0 = (const __bf16*)&nb0;
      const __bf16* e1 = (const __bf16*)&nb1;
      #pragma unroll
      for (int j = 0; j < 8; ++j) {
        sB[nxt][(bn16 + j)     * GLD + bk] = e0[j];
        sB[nxt][(bn16 + 8 + j) * GLD + bk] = e1[j];
      }
      wait_async0();   // next-stage A tile resident before barrier
    }
    __syncthreads();
    cur = nxt;
  }

  // ---- epilogue
  #pragma unroll
  for (int nt = 0; nt < 2; ++nt) {
    const int col = bn + wn * 32 + nt * 16 + l16;
    const float bb = bias[col];
    #pragma unroll
    for (int mt = 0; mt < 2; ++mt) {
      #pragma unroll
      for (int r = 0; r < 8; ++r) {
        const int row = bm + wm * 32 + mt * 16 + r + 8 * half;
        float v = acc[mt][nt][r] + bb;
        const size_t idx = (size_t)row * N + col;
        if (EPI == 0) {
          ((__bf16*)Cout)[idx] = f32_to_bf16(v);
        } else if (EPI == 1) {
          float t = tanhf(0.7978845608028654f * (v + 0.044715f * v * v * v));
          ((__bf16*)Cout)[idx] = f32_to_bf16(0.5f * v * (1.0f + t));
        } else {
          float* O = (float*)Cout;
          O[idx] += v;
        }
      }
    }
  }
}

// ---------------------------------------------------------------- Flash attention
// grid (S/64, NH, B), 128 threads = 4 waves; each wave owns 16 query rows.
#define ALD 72   // padded LDS stride (144B: 16B-aligned, bank-rotating)

__global__ __launch_bounds__(128)
void attn_kernel(const __bf16* __restrict__ qkv, __bf16* __restrict__ out)
{
  __shared__ __align__(16) __bf16 sQ[64 * ALD];
  __shared__ __align__(16) __bf16 sK[64 * ALD];
  __shared__ __align__(16) __bf16 sV[64 * ALD];      // transposed: [dk][key]
  __shared__ __align__(16) __bf16 sP[4][16 * ALD];   // per-wave P staging

  const int tid  = threadIdx.x;
  const int lane = tid & 31;
  const int w    = tid >> 5;      // 0..3
  const int half = lane >> 4;
  const int l16  = lane & 15;
  const int b    = blockIdx.z;
  const int hh   = blockIdx.y;
  const int q0   = blockIdx.x * 64;

  const __bf16* qbase = qkv + (size_t)b * S_ * HQKV_ + hh * DK_;
  const __bf16* kbase = qbase + H_;
  const __bf16* vbase = qbase + 2 * H_;

  const int lr = tid >> 1;            // tile row 0..63
  const int lc = (tid & 1) * 32;      // tile col {0,32}

  // Q tile 64x64 via async engine
  async_ldx4_b128(qbase + (size_t)(q0 + lr) * HQKV_ + lc, &sQ[lr * ALD + lc]);
  wait_async0();
  __syncthreads();

  // hoist Q fragments (sQ is never overwritten)
  FragAB qa[2];
  #pragma unroll
  for (int ks = 0; ks < 2; ++ks) {
    const __bf16* qp = &sQ[(w * 16 + l16) * ALD + ks * 32 + half * 8];
    qa[ks].q[0] = *(const uint4*)qp;
    qa[ks].q[1] = *(const uint4*)(qp + 16);
  }

  float m_r[8], l_r[8];
  v8f o[4] = {};
  #pragma unroll
  for (int r = 0; r < 8; ++r) { m_r[r] = -1e30f; l_r[r] = 0.f; }

  const int nkt = blockIdx.x + 1;   // causal: key tiles 0..qt
  for (int kt = 0; kt < nkt; ++kt) {
    __syncthreads();                // protect sK/sV from previous iteration's reads
    const int kb = kt * 64;
    // K tile 64x64 row-major [key][dk] via async engine
    async_ldx4_b128(kbase + (size_t)(kb + lr) * HQKV_ + lc, &sK[lr * ALD + lc]);
    { // V tile stored transposed [dk][key] (needs VGPR path)
      const uint4* src = (const uint4*)(vbase + (size_t)(kb + lr) * HQKV_ + lc);
      uint4 tmp[4]; tmp[0] = src[0]; tmp[1] = src[1]; tmp[2] = src[2]; tmp[3] = src[3];
      const __bf16* e = (const __bf16*)tmp;
      #pragma unroll
      for (int j = 0; j < 32; ++j) sV[(lc + j) * ALD + lr] = e[j];
    }
    wait_async0();
    __syncthreads();

    // S = Q K^T  (wave: 16 rows x 64 keys = 4 tiles, K-dim 64 = 2 wmma each)
    v8f sacc[4] = {};
    #pragma unroll
    for (int ks = 0; ks < 2; ++ks) {
      #pragma unroll
      for (int nt = 0; nt < 4; ++nt) {
        FragAB kf;
        const __bf16* kp = &sK[(nt * 16 + l16) * ALD + ks * 32 + half * 8];
        kf.q[0] = *(const uint4*)kp;
        kf.q[1] = *(const uint4*)(kp + 16);
        sacc[nt] = __builtin_amdgcn_wmma_f32_16x16x32_bf16(
            false, qa[ks].v, false, kf.v, (short)0, sacc[nt], false, false);
      }
    }

    // online softmax update (row owned by VGPR slot r + lane-half)
    #pragma unroll
    for (int r = 0; r < 8; ++r) {
      const int qrow = q0 + w * 16 + r + 8 * half;
      float mx = -1e30f;
      #pragma unroll
      for (int nt = 0; nt < 4; ++nt) {
        int key = kb + nt * 16 + l16;
        float v = sacc[nt][r] * 0.125f;          // 1/sqrt(64)
        if (key > qrow) v = -1e30f;              // causal mask
        sacc[nt][r] = v;
        mx = fmaxf(mx, v);
      }
      #pragma unroll
      for (int d = 1; d < 16; d <<= 1) mx = fmaxf(mx, __shfl_xor(mx, d, 32));
      float mnew  = fmaxf(m_r[r], mx);
      float alpha = __expf(m_r[r] - mnew);
      float psum  = 0.f;
      #pragma unroll
      for (int nt = 0; nt < 4; ++nt) {
        float p = __expf(sacc[nt][r] - mnew);
        psum += p;
        sP[w][(r + 8 * half) * ALD + nt * 16 + l16] = f32_to_bf16(p);
      }
      #pragma unroll
      for (int d = 1; d < 16; d <<= 1) psum += __shfl_xor(psum, d, 32);
      l_r[r] = l_r[r] * alpha + psum;
      m_r[r] = mnew;
      #pragma unroll
      for (int nt = 0; nt < 4; ++nt) o[nt][r] *= alpha;
    }
    __syncthreads();

    // O += P V   (A = P from per-wave LDS, B = V^T tiles)
    #pragma unroll
    for (int ks = 0; ks < 2; ++ks) {
      FragAB pa;
      const __bf16* pp = &sP[w][l16 * ALD + ks * 32 + half * 8];
      pa.q[0] = *(const uint4*)pp;
      pa.q[1] = *(const uint4*)(pp + 16);
      #pragma unroll
      for (int nt = 0; nt < 4; ++nt) {
        FragAB vf;
        const __bf16* vp = &sV[(nt * 16 + l16) * ALD + ks * 32 + half * 8];
        vf.q[0] = *(const uint4*)vp;
        vf.q[1] = *(const uint4*)(vp + 16);
        o[nt] = __builtin_amdgcn_wmma_f32_16x16x32_bf16(
            false, pa.v, false, vf.v, (short)0, o[nt], false, false);
      }
    }
  }

  // normalize and store merged heads: out[b][row][hh*64 + dk]
  __bf16* obase = out + (size_t)b * S_ * H_ + hh * DK_;
  #pragma unroll
  for (int r = 0; r < 8; ++r) {
    const float inv = 1.0f / l_r[r];
    const int row = q0 + w * 16 + r + 8 * half;
    #pragma unroll
    for (int nt = 0; nt < 4; ++nt)
      obase[(size_t)row * H_ + nt * 16 + l16] = f32_to_bf16(o[nt][r] * inv);
  }
}

// ---------------------------------------------------------------- host launch
extern "C" void kernel_launch(void* const* d_in, const int* in_sizes, int n_in,
                              void* d_out, int out_size, void* d_ws, size_t ws_size,
                              hipStream_t stream)
{
  const float* hidden = (const float*)d_in[0];
  const float* attn_w = (const float*)d_in[1];
  const float* attn_b = (const float*)d_in[2];
  const float* proj_w = (const float*)d_in[3];
  const float* proj_b = (const float*)d_in[4];
  const float* fc_w   = (const float*)d_in[5];
  const float* fc_b   = (const float*)d_in[6];
  const float* fc2_w  = (const float*)d_in[7];
  const float* fc2_b  = (const float*)d_in[8];
  const float* ln1_g  = (const float*)d_in[9];
  const float* ln1_b  = (const float*)d_in[10];
  const float* ln2_g  = (const float*)d_in[11];
  const float* ln2_b  = (const float*)d_in[12];
  const int*   pos    = (const int*)d_in[13];

  char* ws = (char*)d_ws;
  size_t off = 0;
  auto carve = [&](size_t bytes) -> char* {
    char* p = ws + off; off += (bytes + 255) & ~(size_t)255; return p;
  };
  __bf16* w_attn = (__bf16*)carve((size_t)L_ * H_ * HQKV_ * 2);
  __bf16* w_proj = (__bf16*)carve((size_t)L_ * H_ * H_ * 2);
  __bf16* w_fc   = (__bf16*)carve((size_t)L_ * H_ * MLP_ * 2);
  __bf16* w_fc2  = (__bf16*)carve((size_t)L_ * MLP_ * H_ * 2);
  __bf16* xln    = (__bf16*)carve((size_t)MTOT_ * H_ * 2);
  __bf16* qkv    = (__bf16*)carve((size_t)MTOT_ * HQKV_ * 2);
  __bf16* attn_o = (__bf16*)carve((size_t)MTOT_ * H_ * 2);
  __bf16* mid    = (__bf16*)carve((size_t)MTOT_ * MLP_ * 2);
  float*  h      = (float*)d_out;     // running hidden state / residual accumulator

  // per-call weight conversion fp32 -> bf16 (weights then live in L2)
  k_f32_to_bf16<<<dim3((size_t)L_ * H_ * HQKV_ / 1024), 256, 0, stream>>>(attn_w, w_attn);
  k_f32_to_bf16<<<dim3((size_t)L_ * H_ * H_    / 1024), 256, 0, stream>>>(proj_w, w_proj);
  k_f32_to_bf16<<<dim3((size_t)L_ * H_ * MLP_  / 1024), 256, 0, stream>>>(fc_w,  w_fc);
  k_f32_to_bf16<<<dim3((size_t)L_ * MLP_ * H_  / 1024), 256, 0, stream>>>(fc2_w, w_fc2);

  // h = hidden_states
  k_copy_f32<<<dim3((size_t)MTOT_ * H_ / 1024), 256, 0, stream>>>(hidden, h);

  for (int i = 0; i < L_; ++i) {
    // attention block
    ln_kernel<<<MTOT_, 256, 0, stream>>>(h, ln1_g + i * H_, ln1_b + i * H_, xln);
    gemm_bf16<0><<<dim3(HQKV_ / 64, MTOT_ / 128), 256, 0, stream>>>(
        xln, w_attn + (size_t)i * H_ * HQKV_, attn_b + i * HQKV_, qkv, HQKV_, H_);
    rope_kernel<<<dim3((size_t)B_ * S_ * 2 * NH_ * 32 / 256), 256, 0, stream>>>(qkv, pos);
    attn_kernel<<<dim3(S_ / 64, NH_, B_), 128, 0, stream>>>(qkv, attn_o);
    gemm_bf16<2><<<dim3(H_ / 64, MTOT_ / 128), 256, 0, stream>>>(
        attn_o, w_proj + (size_t)i * H_ * H_, proj_b + i * H_, h, H_, H_);
    // MLP block
    ln_kernel<<<MTOT_, 256, 0, stream>>>(h, ln2_g + i * H_, ln2_b + i * H_, xln);
    gemm_bf16<1><<<dim3(MLP_ / 64, MTOT_ / 128), 256, 0, stream>>>(
        xln, w_fc + (size_t)i * H_ * MLP_, fc_b + i * MLP_, mid, MLP_, H_);
    gemm_bf16<2><<<dim3(H_ / 64, MTOT_ / 128), 256, 0, stream>>>(
        mid, w_fc2 + (size_t)i * MLP_ * H_, fc2_b + i * H_, h, H_, MLP_);
  }
}
